// SelfAttentionPooling_19009525252336
// MI455X (gfx1250) — matrix operand order repaired and checked by reading, not compiled, not run on hardware
//
#include <hip/hip_runtime.h>

// ---------------------------------------------------------------------------
// Fused self-attention pooling for MI455X (gfx1250, wave32, WMMA).
// One workgroup (256 thr = 8 waves) per batch element; all intermediates in
// LDS (~207KB of the 320KB/WGP) so no HBM round-trip for scores/attn.
//   phase 0: x = input + pos_enc -> LDS f16; Wq/Wk/Wv -> LDS f16; row mask
//   phase 1: q/k/v = x @ W.T + b via v_wmma_f32_16x16x32_f16 (q pre-scaled)
//   phase 2: per (head, row-tile) flash-attention over 7 chunks of 32 keys
//            (2 score WMMAs + 1 P.V WMMA per chunk), key-side mask only
//   phase 3: masked row-sum, then tiny 64x64 GEMV with Wf + bf  -> d_out
// V is stored column-major [h][d][l] so both its WMMA B-gather and its WMMA
// C-store are contiguous b128 LDS ops.
// ---------------------------------------------------------------------------

typedef __attribute__((ext_vector_type(16))) _Float16 v16h;
typedef __attribute__((ext_vector_type(8)))  _Float16 v8h;
typedef __attribute__((ext_vector_type(4)))  _Float16 v4h;
typedef __attribute__((ext_vector_type(8)))  float    v8f;

#define BSZ   512
#define LSEQ  200
#define EDIM  64
#define HH    4
#define LP    224     // L padded to 14 tiles of 16 (= 7 chunks of 32)
#define NT    14
#define NC    7
#define NEGF  (-1.0e30f)

// LDS layout (bytes)
#define OFF_XS    0        // _Float16 [LP][64]    x, later reused as attn out
#define OFF_QS    28672    // _Float16 [4][LP][32] (D padded to 32, upper = 0)
#define OFF_KS    86016    // _Float16 [4][LP][32]
#define OFF_VS    143360   // _Float16 [4][16][LP] column-major V
#define OFF_PSC   172032   // _Float16 [8][16*32]  per-wave P-tile scratch
#define OFF_WSH   180224   // _Float16 [3][64][64] f16 copies of Wq/Wk/Wv
#define OFF_RM    204800   // int      [LP]
#define OFF_PART  205696   // float    [4][64]
#define OFF_LEN   206720   // float    [1]
#define LDS_BYTES 206848

__device__ __forceinline__ v8f wmma_f16(v16h a, v16h b, v8f c) {
  return __builtin_amdgcn_wmma_f32_16x16x32_f16(false, a, false, b,
                                                (short)0, c, false, false);
}

__device__ __forceinline__ v16h cat8(v8h a, v8h b) {
  v16h r;
  #pragma unroll
  for (int j = 0; j < 8; ++j) { r[j] = a[j]; r[j + 8] = b[j]; }
  return r;
}

extern "C" __global__ __launch_bounds__(256)
void sap_fused_wmma(const float* __restrict__ input,
                    const unsigned char* __restrict__ mask,  // jnp bool = 1B
                    const float* __restrict__ pos_enc,
                    const float* __restrict__ Wk, const float* __restrict__ bk,
                    const float* __restrict__ Wv, const float* __restrict__ bv,
                    const float* __restrict__ Wq, const float* __restrict__ bq,
                    const float* __restrict__ Wf, const float* __restrict__ bf,
                    float* __restrict__ out)
{
  extern __shared__ char lds[];
  _Float16* xs  = (_Float16*)(lds + OFF_XS);
  _Float16* qs  = (_Float16*)(lds + OFF_QS);
  _Float16* ks  = (_Float16*)(lds + OFF_KS);
  _Float16* vs  = (_Float16*)(lds + OFF_VS);   // [h][d][l]
  _Float16* psc = (_Float16*)(lds + OFF_PSC);
  _Float16* wsh = (_Float16*)(lds + OFF_WSH);
  int*   rmask = (int*)(lds + OFF_RM);
  float* part  = (float*)(lds + OFF_PART);
  float* lenp  = (float*)(lds + OFF_LEN);

  const int b    = blockIdx.x;
  const int tid  = threadIdx.x;
  const int lane = tid & 31;
  const int wave = tid >> 5;
  const int lo   = lane & 15;
  const int hi   = lane >> 4;   // 0 | 1
  const int hi8  = hi * 8;
  const int hi16 = hi * 16;

  // ---------------- phase 0: stage x (f16), weights (f16), mask --------------
  for (int i = tid; i < (LP * EDIM) / 8; i += 256) {   // 8 halves per chunk
    const int l = i >> 3;
    const int e = (i & 7) * 8;
    v8h hv;
    if (l < LSEQ) {
      const float4* pi = (const float4*)(input + ((long)b * LSEQ + l) * EDIM + e);
      const float4* pp = (const float4*)(pos_enc + l * EDIM + e);
      const float4 x0 = pi[0], x1 = pi[1];
      const float4 p0 = pp[0], p1 = pp[1];
      hv[0] = (_Float16)(x0.x + p0.x); hv[1] = (_Float16)(x0.y + p0.y);
      hv[2] = (_Float16)(x0.z + p0.z); hv[3] = (_Float16)(x0.w + p0.w);
      hv[4] = (_Float16)(x1.x + p1.x); hv[5] = (_Float16)(x1.y + p1.y);
      hv[6] = (_Float16)(x1.z + p1.z); hv[7] = (_Float16)(x1.w + p1.w);
    } else {
      #pragma unroll
      for (int j = 0; j < 8; ++j) hv[j] = (_Float16)0.0f;
    }
    *(v8h*)(xs + i * 8) = hv;
  }
  {
    v8h z;
    #pragma unroll
    for (int j = 0; j < 8; ++j) z[j] = (_Float16)0.0f;
    for (int i = tid; i < (HH * LP * 32) / 8; i += 256) { // zero (D-pad cols)
      *(v8h*)(qs + i * 8) = z;
      *(v8h*)(ks + i * 8) = z;
    }
  }
  #pragma unroll
  for (int mm = 0; mm < 3; ++mm) {                       // weights -> LDS f16
    const float* W = (mm == 0) ? Wq : (mm == 1 ? Wk : Wv);
    _Float16* dst = wsh + mm * 4096;
    for (int i = tid * 4; i < 4096; i += 1024) {
      const float4 w = *(const float4*)(W + i);
      v4h h4;
      h4[0] = (_Float16)w.x; h4[1] = (_Float16)w.y;
      h4[2] = (_Float16)w.z; h4[3] = (_Float16)w.w;
      *(v4h*)(dst + i) = h4;
    }
  }
  for (int i = tid; i < LP; i += 256)
    rmask[i] = (i < LSEQ) ? (mask[(long)b * LSEQ + i] ? 1 : 0) : 0;
  __syncthreads();

  // ---------------- phase 1: q/k/v projections (WMMA) ----------------
  // out[n,e] = sum_j x[n,j] * W[e,j] + bias[e]   (q additionally * 0.25)
  for (int task = wave; task < 3 * NT * HH; task += 8) {
    const int mat = task / (NT * HH);            // 0=q 1=k 2=v
    const int rem = task - mat * (NT * HH);
    const int it  = rem >> 2;                    // row tile
    const int h   = rem & 3;                     // e-tile == head
    const float* bias = (mat == 0) ? bq : (mat == 1 ? bk : bv);
    const int arow = it * 16 + lo;               // A-matrix row for this lane
    v8f acc = {};
    #pragma unroll
    for (int kc = 0; kc < 2; ++kc) {             // K = 64 in two chunks of 32
      const _Float16* ab = xs + arow * 64 + kc * 32 + hi16;
      const v16h a = cat8(*(const v8h*)ab, *(const v8h*)(ab + 8));
      const _Float16* wb = wsh + mat * 4096 + (h * 16 + lo) * 64 + kc * 32 + hi16;
      const v16h bm = cat8(*(const v8h*)wb, *(const v8h*)(wb + 8));
      acc = wmma_f16(a, bm, acc);
    }
    const float bb = bias[h * 16 + lo];
    #pragma unroll
    for (int r = 0; r < 8; ++r) acc[r] += bb;
    if (mat == 0) {
      #pragma unroll
      for (int r = 0; r < 8; ++r) acc[r] *= 0.25f;   // 1/sqrt(D)
    }
    if (mat == 2) {                              // V column-major: 1 b128 store
      v8h hv;
      #pragma unroll
      for (int r = 0; r < 8; ++r) hv[r] = (_Float16)acc[r];
      *(v8h*)(vs + (h * 16 + lo) * LP + it * 16 + hi8) = hv;
    } else {
      #pragma unroll
      for (int r = 0; r < 8; ++r) {              // C layout: M = r + 8*hi
        const int orow = it * 16 + r + hi8;
        if (mat == 0) qs[(h * LP + orow) * 32 + lo] = (_Float16)acc[r];
        else          ks[(h * LP + orow) * 32 + lo] = (_Float16)acc[r];
      }
    }
  }
  __syncthreads();

  // ---------------- phase 2: flash attention per (head, row tile) ------------
  _Float16* pw = psc + wave * (16 * 32);
  for (int task = wave; task < HH * NT; task += 8) {
    const int h  = task / NT;
    const int it = task - h * NT;
    const int qrow = it * 16 + lo;
    const _Float16* qb = qs + (h * LP + qrow) * 32 + hi16;
    const v16h aq = cat8(*(const v8h*)qb, *(const v8h*)(qb + 8));  // 16x32, hi K=0
    float m[8], den[8];
    v8f O = {};
    #pragma unroll
    for (int r = 0; r < 8; ++r) { m[r] = NEGF; den[r] = 0.0f; }

    for (int c = 0; c < NC; ++c) {
      const int kb = c * 32;
      const _Float16* k0 = ks + (h * LP + kb + lo) * 32 + hi16;
      const _Float16* k1 = ks + (h * LP + kb + 16 + lo) * 32 + hi16;
      const v16h b0 = cat8(*(const v8h*)k0, *(const v8h*)(k0 + 8));
      const v16h b1 = cat8(*(const v8h*)k1, *(const v8h*)(k1 + 8));
      v8f z = {};
      v8f S0 = wmma_f16(aq, b0, z);              // cols kb..kb+15
      v8f S1 = wmma_f16(aq, b1, z);              // cols kb+16..kb+31

      const bool v0 = rmask[kb + lo] != 0;       // key-side mask (per column)
      const bool v1 = rmask[kb + 16 + lo] != 0;
      float cmx[8];
      #pragma unroll
      for (int r = 0; r < 8; ++r) {
        const float s0 = v0 ? S0[r] : NEGF;
        const float s1 = v1 ? S1[r] : NEGF;
        S0[r] = s0; S1[r] = s1;
        cmx[r] = fmaxf(s0, s1);
      }
      #pragma unroll
      for (int sh = 1; sh < 16; sh <<= 1) {      // row max over 16-lane group
        #pragma unroll
        for (int r = 0; r < 8; ++r)
          cmx[r] = fmaxf(cmx[r], __shfl_xor(cmx[r], sh, 32));
      }
      float p0[8], p1[8], rs[8];
      #pragma unroll
      for (int r = 0; r < 8; ++r) {
        const float mn = fmaxf(m[r], cmx[r]);
        const float sc = __expf(m[r] - mn);
        m[r] = mn;
        p0[r] = __expf(S0[r] - mn);
        p1[r] = __expf(S1[r] - mn);
        rs[r] = p0[r] + p1[r];
        den[r] *= sc;
        O[r] *= sc;
      }
      #pragma unroll
      for (int sh = 1; sh < 16; sh <<= 1) {      // row sum over 16-lane group
        #pragma unroll
        for (int r = 0; r < 8; ++r)
          rs[r] += __shfl_xor(rs[r], sh, 32);
      }
      #pragma unroll
      for (int r = 0; r < 8; ++r) den[r] += rs[r];

      // C-layout -> A-layout for P via per-wave LDS scratch
      #pragma unroll
      for (int r = 0; r < 8; ++r) {
        pw[(r + hi8) * 32 + lo]      = (_Float16)p0[r];
        pw[(r + hi8) * 32 + 16 + lo] = (_Float16)p1[r];
      }
      asm volatile("s_wait_dscnt 0" ::: "memory");  // same-wave LDS RAW
      const _Float16* pb = pw + lo * 32 + hi16;
      const v16h ap = cat8(*(const v8h*)pb, *(const v8h*)(pb + 8));
      const _Float16* vb = vs + (h * 16 + lo) * LP + kb + hi16;  // column-major
      const v16h bvv = cat8(*(const v8h*)vb, *(const v8h*)(vb + 8));
      O = wmma_f16(ap, bvv, O);                  // O += P * V, K = 32 keys
    }
    #pragma unroll
    for (int r = 0; r < 8; ++r) {                // normalize, merge heads (xs
      const float o = O[r] / (den[r] + 1e-20f);  //  is dead -> reuse as "os")
      xs[(it * 16 + r + hi8) * 64 + h * 16 + lo] = (_Float16)o;
    }
  }
  __syncthreads();

  // ---------------- phase 3: masked mean-pool then Wf GEMV -------------------
  // pooled = ((1/len) * sum_{valid l} os[l,:]) @ Wf.T + bf   (linearity)
  if (tid == 0) {
    int n = 0;
    for (int i = 0; i < LSEQ; ++i) n += rmask[i];
    lenp[0] = (float)n;
  }
  __syncthreads();
  {
    const int col = tid & 63, g = tid >> 6;
    float p = 0.0f;
    for (int l = g; l < LP; l += 4)
      if (rmask[l]) p += (float)xs[l * 64 + col];
    part[g * 64 + col] = p;
  }
  __syncthreads();
  if (tid < 64) {
    const float s = (part[tid] + part[64 + tid] + part[128 + tid] +
                     part[192 + tid]) / (lenp[0] + 1e-8f);
    part[tid] = s;
  }
  __syncthreads();
  if (tid < 64) {
    float acc = bf[tid];
    for (int j = 0; j < 64; ++j) acc += part[j] * Wf[tid * 64 + j];
    out[(long)b * 64 + tid] = acc;
  }
}

extern "C" void kernel_launch(void* const* d_in, const int* in_sizes, int n_in,
                              void* d_out, int out_size, void* d_ws, size_t ws_size,
                              hipStream_t stream) {
  const float*         input   = (const float*)d_in[0];
  const unsigned char* mask    = (const unsigned char*)d_in[1];
  const float*         pos_enc = (const float*)d_in[2];
  const float*         Wk      = (const float*)d_in[3];
  const float*         bk      = (const float*)d_in[4];
  const float*         Wv      = (const float*)d_in[5];
  const float*         bv      = (const float*)d_in[6];
  const float*         Wq      = (const float*)d_in[7];
  const float*         bq      = (const float*)d_in[8];
  const float*         Wf      = (const float*)d_in[9];
  const float*         bf      = (const float*)d_in[10];
  sap_fused_wmma<<<BSZ, 256, LDS_BYTES, stream>>>(
      input, mask, pos_enc, Wk, bk, Wv, bv, Wq, bq, Wf, bf, (float*)d_out);
}
